// Net_70858370449981
// MI455X (gfx1250) — matrix operand order repaired and checked by reading, not compiled, not run on hardware
//
#include <hip/hip_runtime.h>
#include <hip/hip_bf16.h>

typedef __attribute__((ext_vector_type(16))) _Float16 v16h;
typedef __attribute__((ext_vector_type(8)))  _Float16 v8h;
typedef __attribute__((ext_vector_type(8)))  float    v8f;

#define NODES   50000
#define IN_DIM  128
#define HID_DIM 64
#define POS_DIM 32
#define E_MP    800000
#define E_LP    100000

__device__ __forceinline__ float wave_reduce_add(float v) {
    #pragma unroll
    for (int off = 16; off > 0; off >>= 1)
        v += __shfl_xor(v, off, 32);
    return v;
}

// ---------------- degree / normalization ----------------
__global__ __launch_bounds__(256) void deg_init(float* __restrict__ deg, int n) {
    int i = blockIdx.x * 256 + threadIdx.x;
    if (i < n) deg[i] = 1.0f;                       // self-loop contributes 1
}

__global__ __launch_bounds__(256) void deg_count(const int* __restrict__ col,
                                                 float* __restrict__ deg, int e) {
    int i = blockIdx.x * 256 + threadIdx.x;
    if (i < e) atomicAdd(&deg[col[i]], 1.0f);
}

__global__ __launch_bounds__(256) void deg_rsqrt(float* __restrict__ deg, int n) {
    int i = blockIdx.x * 256 + threadIdx.x;
    if (i < n) {
        float d = deg[i];
        deg[i] = (d > 0.0f) ? rsqrtf(d) : 0.0f;     // becomes dinv in place
    }
}

// ---------------- WMMA GEMM: G[M x 64] = A[M x K] @ W[K x 64] ----------------
// 128 threads = 4 waves; block owns 16 rows, wave w owns cols [16w,16w+16).
__global__ __launch_bounds__(128) void gemm_wmma(const float* __restrict__ A,
                                                 const float* __restrict__ W,
                                                 float* __restrict__ G,
                                                 int M, int K) {
    __shared__ _Float16 sA[16 * 128];     // [row][k], stride K
    __shared__ _Float16 sBt[64 * 128];    // W transposed: [n][k], stride K

    const int tid  = threadIdx.x;
    const int row0 = blockIdx.x * 16;

    // stage A tile (f32 -> f16)
    for (int idx = tid; idx < 16 * K; idx += 128) {
        int r = idx / K, k = idx - r * K;
        int gr = row0 + r;
        float v = (gr < M) ? A[(size_t)gr * K + k] : 0.0f;
        sA[r * K + k] = (_Float16)v;
    }
    // stage W transposed (f32 -> f16)
    for (int idx = tid; idx < K * 64; idx += 128) {
        int k = idx >> 6, n = idx & 63;
        sBt[n * K + k] = (_Float16)W[idx];
    }
    __syncthreads();

    const int wave = tid >> 5;
    const int lane = tid & 31;
    const int n0   = wave * 16;
    const int m    = lane & 15;          // A row within tile / D col
    const int half = lane >> 4;

    v8f acc = {};
    for (int k0 = 0; k0 < K; k0 += 32) {
        // A fragment: lane half 0 -> K {0..7,16..23}; half 1 -> K {8..15,24..31}
        int abase = m * K + k0 + half * 8;
        v8h alo = *(const v8h*)&sA[abase];
        v8h ahi = *(const v8h*)&sA[abase + 16];
        v16h a = __builtin_shufflevector(alo, ahi,
                 0,1,2,3,4,5,6,7,8,9,10,11,12,13,14,15);
        // B fragment: lane n = lane%16; half 0 -> K {0..7,16..23}; half 1 -> K {8..15,24..31}
        int bbase = (n0 + m) * K + k0 + half * 8;
        v8h blo = *(const v8h*)&sBt[bbase];
        v8h bhi = *(const v8h*)&sBt[bbase + 16];
        v16h b = __builtin_shufflevector(blo, bhi,
                 0,1,2,3,4,5,6,7,8,9,10,11,12,13,14,15);
        acc = __builtin_amdgcn_wmma_f32_16x16x32_f16(
                  false, a, false, b, (short)0, acc, false, false);
    }

    // D layout: VGPR r -> M = r + 8*half, N = lane%16
    const int n     = n0 + m;
    const int mbase = row0 + half * 8;
    #pragma unroll
    for (int r = 0; r < 8; ++r) {
        int gm = mbase + r;
        if (gm < M) G[(size_t)gm * 64 + n] = acc[r];
    }
}

// ---------------- H init with bias ----------------
__global__ __launch_bounds__(256) void bias_init(float* __restrict__ H,
                                                 const float* __restrict__ b, int n) {
    int i = blockIdx.x * 256 + threadIdx.x;
    if (i < n * HID_DIM) H[i] = b[i & (HID_DIM - 1)];
}

// ---------------- gated message passing (one wave32 per edge) ----------------
__global__ __launch_bounds__(256) void peg_aggregate(const float* __restrict__ G,
                                                     const int*   __restrict__ tp,
                                                     const float* __restrict__ pos,
                                                     const float* __restrict__ dinv,
                                                     const float* __restrict__ em_w,
                                                     const float* __restrict__ em_b,
                                                     float* __restrict__ H,
                                                     int n, int e) {
    const int wave = threadIdx.x >> 5;
    const int lane = threadIdx.x & 31;
    const int edge = blockIdx.x * 8 + wave;
    const int etot = e + n;                          // train edges + self loops
    if (edge >= etot) return;

    int r, c;
    float sq = 0.0f;
    if (edge < e) {
        r = tp[edge];
        c = tp[e + edge];
        float dd = pos[(size_t)r * POS_DIM + lane] - pos[(size_t)c * POS_DIM + lane];
        sq = wave_reduce_add(dd * dd);
    } else {
        r = c = edge - e;                            // self loop: sq = 0
    }
    float z    = sq * em_w[0] + em_b[0];
    float gate = 1.0f / (1.0f + __expf(-z));
    float coef = gate * dinv[r] * dinv[c];

    float m0 = coef * G[(size_t)r * 64 + lane];
    float m1 = coef * G[(size_t)r * 64 + 32 + lane];
    atomicAdd(&H[(size_t)c * 64 + lane],      m0);
    atomicAdd(&H[(size_t)c * 64 + 32 + lane], m1);
}

// ---------------- final edge scoring (one wave32 per eval edge) ----------------
__global__ __launch_bounds__(256) void edge_score(const float* __restrict__ H,
                                                  const float* __restrict__ pos,
                                                  const int*   __restrict__ pe,
                                                  const int*   __restrict__ ne,
                                                  const float* __restrict__ fc_w,
                                                  const float* __restrict__ fc_b,
                                                  float* __restrict__ out, int el) {
    const int wave = threadIdx.x >> 5;
    const int lane = threadIdx.x & 31;
    const int e = blockIdx.x * 8 + wave;
    if (e >= 2 * el) return;

    int s, d;
    if (e < el) { s = pe[e];      d = pe[el + e];      }
    else        { s = ne[e - el]; d = ne[el + e - el]; }

    float acc = H[(size_t)s * 64 + lane]      * H[(size_t)d * 64 + lane]
              + H[(size_t)s * 64 + 32 + lane] * H[(size_t)d * 64 + 32 + lane];
    float pp  = pos[(size_t)s * POS_DIM + lane] - pos[(size_t)d * POS_DIM + lane];
    float dot = wave_reduce_add(acc);
    float sq  = wave_reduce_add(pp * pp);

    if (lane == 0) out[e] = dot * fc_w[0] + sq * fc_w[1] + fc_b[0];
}

extern "C" void kernel_launch(void* const* d_in, const int* in_sizes, int n_in,
                              void* d_out, int out_size, void* d_ws, size_t ws_size,
                              hipStream_t stream) {
    const float* x     = (const float*)d_in[0];
    const float* pos   = (const float*)d_in[1];
    const int*   pe    = (const int*)  d_in[2];
    const int*   ne    = (const int*)  d_in[3];
    const int*   tp    = (const int*)  d_in[4];
    const float* W1    = (const float*)d_in[5];
    const float* b1    = (const float*)d_in[6];
    const float* em1w  = (const float*)d_in[7];
    const float* em1b  = (const float*)d_in[8];
    const float* W2    = (const float*)d_in[9];
    const float* b2    = (const float*)d_in[10];
    const float* em2w  = (const float*)d_in[11];
    const float* em2b  = (const float*)d_in[12];
    const float* fcw   = (const float*)d_in[13];
    const float* fcb   = (const float*)d_in[14];
    float*       out   = (float*)d_out;

    // workspace carve-up
    char* ws = (char*)d_ws;
    float* dinv = (float*)(ws);                                   // 0.2 MB
    float* G    = (float*)(ws + (size_t)(1  << 20));              // 12.8 MB
    float* H1   = (float*)(ws + (size_t)(15 << 20));              // 12.8 MB
    float* H2   = (float*)(ws + (size_t)(29 << 20));              // 12.8 MB

    const int N = NODES;
    const int gemm_blocks = (N + 15) / 16;                        // 3125
    const int et1 = E_MP + N;

    // GCN normalization
    deg_init <<<(N + 255) / 256, 256, 0, stream>>>(dinv, N);
    deg_count<<<(E_MP + 255) / 256, 256, 0, stream>>>(tp + E_MP, dinv, E_MP);
    deg_rsqrt<<<(N + 255) / 256, 256, 0, stream>>>(dinv, N);

    // layer 1: G = x @ W1 ; H1 = b1 + scatter(gate*norm*G)
    gemm_wmma<<<gemm_blocks, 128, 0, stream>>>(x, W1, G, N, IN_DIM);
    bias_init<<<(N * HID_DIM + 255) / 256, 256, 0, stream>>>(H1, b1, N);
    peg_aggregate<<<(et1 + 7) / 8, 256, 0, stream>>>(G, tp, pos, dinv, em1w, em1b, H1, N, E_MP);

    // layer 2: G = H1 @ W2 ; H2 = b2 + scatter(gate*norm*G)
    gemm_wmma<<<gemm_blocks, 128, 0, stream>>>(H1, W2, G, N, HID_DIM);
    bias_init<<<(N * HID_DIM + 255) / 256, 256, 0, stream>>>(H2, b2, N);
    peg_aggregate<<<(et1 + 7) / 8, 256, 0, stream>>>(G, tp, pos, dinv, em2w, em2b, H2, N, E_MP);

    // eval-edge scoring
    edge_score<<<(2 * E_LP + 7) / 8, 256, 0, stream>>>(H2, pos, pe, ne, fcw, fcb, out, E_LP);
}
